// LawformerCLSEncoder_44238163148862
// MI455X (gfx1250) — compile-verified, matrix operand order checked by reading, not run on hardware
//
#include <hip/hip_runtime.h>
#include <hip/hip_bf16.h>

// ---------------- problem constants (match reference) ----------------
static constexpr int Bc  = 2;
static constexpr int Sc  = 4096;
static constexpr int Dc  = 768;
static constexpr int Hc  = 12;
static constexpr int DHc = 64;
static constexpr int Wc  = 256;
static constexpr int Lc  = 2;
static constexpr int Gc  = 64;
static constexpr int FFc = 3072;
static constexpr float NEGF = -1.0e9f;

typedef __attribute__((ext_vector_type(16))) _Float16 v16h;
typedef __attribute__((ext_vector_type(8)))  _Float16 v8h;
typedef __attribute__((ext_vector_type(4)))  _Float16 v4h;
typedef __attribute__((ext_vector_type(2)))  _Float16 v2h;
typedef __attribute__((ext_vector_type(8)))  float    v8f;

// ---------------- CDNA5 async global->LDS copy (16B per lane) ----------------
// GLOBAL_LOAD_ASYNC_TO_LDS_B128: per-lane 16B copy from global vaddr to LDS address vdst, tracked by
// ASYNCcnt (ISA 15.18.3 opcode 98). LDS byte address = low 32 bits of the
// generic shared pointer (ISA 10.2).
__device__ __forceinline__ void async_copy16(uint32_t lds_addr,
                                             const _Float16* gsrc) {
  asm volatile("global_load_async_to_lds_b128 %0, %1, off"
               :: "v"(lds_addr), "v"(gsrc) : "memory");
}
__device__ __forceinline__ void wait_async0() {
  asm volatile("s_wait_asynccnt 0x0" ::: "memory");
}

// ---------------- WMMA fragment loader (wave32, 16x16x32 f16) ----------------
// lane&15 = row, lane>>4 selects K-halves; two contiguous 16B spans per lane
// -> 2x ds_load_b128. ld must be a multiple of 8 halves (16B rows).
__device__ __forceinline__ v16h frag_ld(const _Float16* lds, int ld, int row0,
                                        int k0, int lane) {
  int r = lane & 15, sub = (lane >> 4) & 1;
  const _Float16* p = lds + (size_t)(row0 + r) * ld + k0 + sub * 8;
  v8h lo = *(const v8h*)(p);
  v8h hi = *(const v8h*)(p + 16);
  return __builtin_shufflevector(lo, hi, 0, 1, 2, 3, 4, 5, 6, 7,
                                 8, 9, 10, 11, 12, 13, 14, 15);
}

__device__ __forceinline__ float gelu_exact(float x) {
  return 0.5f * x * (1.0f + erff(x * 0.70710678118654752f));
}

// ---------------- weight convert + transpose: WT[n*K+k] = f16(W[k*N+n]) ------
__global__ void convert_transpose_kernel(const float* __restrict__ W,
                                         _Float16* __restrict__ WT,
                                         int K, int N) {
  size_t i = (size_t)blockIdx.x * blockDim.x + threadIdx.x;
  size_t total = (size_t)K * N;
  if (i < total) {
    size_t n = i / K, k = i % K;
    WT[i] = (_Float16)W[k * (size_t)N + n];
  }
}

// ---------------- WMMA GEMM, all-f16 operands, async LDS staging -------------
// C32/C16 = act(Ah[M,K] @ BT[N,K]^T + bias (+ res)); block tile 64x128,
// 8 waves (2x4), each wave 2x2 WMMA tiles; A/B tiles staged with
// global_load_async_to_lds_b128 (no VGPR round trip, no conversion VALU).
__global__ __launch_bounds__(256) void wmma_gemm_kernel(
    const _Float16* __restrict__ Ah, const _Float16* __restrict__ BT,
    const float* __restrict__ bias, const float* __restrict__ res,
    float* __restrict__ C32, _Float16* __restrict__ C16,
    int M, int N, int K, int act) {
  __shared__ __align__(16) _Float16 As[64][40];    // [m][k], 80B rows
  __shared__ __align__(16) _Float16 Bs[128][40];   // [n][k], 80B rows
  int tid = threadIdx.x;
  int lane = tid & 31, wave = tid >> 5;
  int wm = wave >> 2, wn = wave & 3;               // 2 x 4 wave grid
  int row0 = blockIdx.y * 64, col0 = blockIdx.x * 128;

  uint32_t ldsA = (uint32_t)(uintptr_t)&As[0][0];
  uint32_t ldsB = (uint32_t)(uintptr_t)&Bs[0][0];

  v8f acc00 = {}, acc01 = {}, acc10 = {}, acc11 = {};
  for (int kk = 0; kk < K; kk += 32) {
    // A tile: 64 rows x 32 halves = 256 x 16B chunks (1 per thread)
    {
      int r = tid >> 2, c8 = (tid & 3) * 8;
      async_copy16(ldsA + (uint32_t)(r * 80 + c8 * 2),
                   &Ah[(size_t)(row0 + r) * K + kk + c8]);
    }
    // B tile: 128 rows x 32 halves = 512 x 16B chunks (2 per thread)
    for (int u = tid; u < 512; u += 256) {
      int n = u >> 2, c8 = (u & 3) * 8;
      async_copy16(ldsB + (uint32_t)(n * 80 + c8 * 2),
                   &BT[(size_t)(col0 + n) * K + kk + c8]);
    }
    wait_async0();
    __syncthreads();
    v16h a0 = frag_ld(&As[0][0], 40, wm * 32, 0, lane);
    v16h a1 = frag_ld(&As[0][0], 40, wm * 32 + 16, 0, lane);
    v16h b0 = frag_ld(&Bs[0][0], 40, wn * 32, 0, lane);
    v16h b1 = frag_ld(&Bs[0][0], 40, wn * 32 + 16, 0, lane);
    acc00 = __builtin_amdgcn_wmma_f32_16x16x32_f16(false, a0, false, b0, (short)0, acc00, false, false);
    acc01 = __builtin_amdgcn_wmma_f32_16x16x32_f16(false, a0, false, b1, (short)0, acc01, false, false);
    acc10 = __builtin_amdgcn_wmma_f32_16x16x32_f16(false, a1, false, b0, (short)0, acc10, false, false);
    acc11 = __builtin_amdgcn_wmma_f32_16x16x32_f16(false, a1, false, b1, (short)0, acc11, false, false);
    __syncthreads();
  }
  int sub = (lane >> 4) & 1, rl = lane & 15;
#pragma unroll
  for (int ti = 0; ti < 2; ++ti) {
#pragma unroll
    for (int tj = 0; tj < 2; ++tj) {
      v8f accv = (ti == 0) ? (tj == 0 ? acc00 : acc01)
                           : (tj == 0 ? acc10 : acc11);
#pragma unroll
      for (int j = 0; j < 8; ++j) {
        int gr = row0 + wm * 32 + ti * 16 + sub * 8 + j;
        int gc = col0 + wn * 32 + tj * 16 + rl;
        float vv = accv[j] + bias[gc];
        if (res) vv += res[(size_t)gr * N + gc];
        if (act == 1) vv = gelu_exact(vv);
        if (C32) C32[(size_t)gr * N + gc] = vv;
        if (C16) C16[(size_t)gr * N + gc] = (_Float16)vv;
      }
    }
  }
}

// ---------------- LayerNorm over D=768, f32 in, f32 + optional f16 out -------
__global__ __launch_bounds__(256) void ln_kernel(const float* in, const float* s,
                                                 const float* b, float* out,
                                                 _Float16* out_h) {
  __shared__ float red[256];
  int row = blockIdx.x, tid = threadIdx.x;
  const float* x = in + (size_t)row * Dc;
  float sum = 0.f;
  for (int i = tid; i < Dc; i += 256) sum += x[i];
  red[tid] = sum; __syncthreads();
  for (int off = 128; off > 0; off >>= 1) {
    if (tid < off) red[tid] += red[tid + off];
    __syncthreads();
  }
  float mean = red[0] * (1.0f / Dc); __syncthreads();
  float vs = 0.f;
  for (int i = tid; i < Dc; i += 256) { float d = x[i] - mean; vs += d * d; }
  red[tid] = vs; __syncthreads();
  for (int off = 128; off > 0; off >>= 1) {
    if (tid < off) red[tid] += red[tid + off];
    __syncthreads();
  }
  float rstd = rsqrtf(red[0] * (1.0f / Dc) + 1e-5f);
  for (int i = tid; i < Dc; i += 256) {
    float vv = (x[i] - mean) * rstd * s[i] + b[i];
    out[(size_t)row * Dc + i] = vv;
    if (out_h) out_h[(size_t)row * Dc + i] = (_Float16)vv;
  }
}

// ---------------- embeddings + LN (f32 + f16 out) ----------------
__global__ __launch_bounds__(256) void embed_kernel(
    const int* __restrict__ ids, const int* __restrict__ tt,
    const float* __restrict__ we, const float* __restrict__ pe,
    const float* __restrict__ te, const float* __restrict__ lns,
    const float* __restrict__ lnb, float* __restrict__ x,
    _Float16* __restrict__ xh) {
  __shared__ float rowbuf[Dc];
  __shared__ float red[256];
  int row = blockIdx.x, tid = threadIdx.x;
  int s = row % Sc;
  int token = ids[row], typ = tt[row];
  float sum = 0.f;
  for (int i = tid; i < Dc; i += 256) {
    float e = we[(size_t)token * Dc + i] + pe[(size_t)s * Dc + i] +
              te[(size_t)typ * Dc + i];
    rowbuf[i] = e;
    sum += e;
  }
  red[tid] = sum; __syncthreads();
  for (int off = 128; off > 0; off >>= 1) {
    if (tid < off) red[tid] += red[tid + off];
    __syncthreads();
  }
  float mean = red[0] * (1.0f / Dc); __syncthreads();
  float vs = 0.f;
  for (int i = tid; i < Dc; i += 256) { float d = rowbuf[i] - mean; vs += d * d; }
  red[tid] = vs; __syncthreads();
  for (int off = 128; off > 0; off >>= 1) {
    if (tid < off) red[tid] += red[tid + off];
    __syncthreads();
  }
  float rstd = rsqrtf(red[0] * (1.0f / Dc) + 1e-5f);
  for (int i = tid; i < Dc; i += 256) {
    float vv = (rowbuf[i] - mean) * rstd * lns[i] + lnb[i];
    x[(size_t)row * Dc + i] = vv;
    xh[(size_t)row * Dc + i] = (_Float16)vv;
  }
}

// ---------------- is_glob scatter ----------------
__global__ void isglob_kernel(const int* cls_ids, const int* cls_len, int* isg) {
  int i = blockIdx.x * blockDim.x + threadIdx.x;
  if (i < Bc * Gc) {
    int b = i / Gc, g = i % Gc;
    if (g < cls_len[b]) atomicOr(&isg[b * Sc + cls_ids[i]], 1);
  }
}

// ---------------- flash attention over f16 Q/K/V -----------------------------
// mode 0: grid (W/32, nc, B*H) — 32 band queries; key tiles = [2 global | 24 band]
// mode 1: grid (G/32, 1,  B*H) — 32 gathered CLS queries over all S keys
__global__ __launch_bounds__(256) void flash_attn_kernel(
    const _Float16* __restrict__ q, const _Float16* __restrict__ k,
    const _Float16* __restrict__ v, const int* __restrict__ am,
    const int* __restrict__ isg, const int* __restrict__ cls_ids,
    const int* __restrict__ cls_len, _Float16* __restrict__ outp, int mode) {
  __shared__ __align__(16) _Float16 qs[32][72];    // [query][dh]
  __shared__ __align__(16) _Float16 ks[32][72];    // [key][dh]  (== B^T for q@k^T)
  __shared__ __align__(16) _Float16 vst[64][40];   // transposed V: [dh][key]
  __shared__ __align__(16) float    sc[32][36];
  __shared__ __align__(16) _Float16 pr[32][40];    // probs [query][key]
  __shared__ float mrow[32], lrow[32], cfrow[32];
  __shared__ int colok[32];

  int tid = threadIdx.x, lane = tid & 31, wave = tid >> 5;
  int bh = blockIdx.z, b = bh / Hc, h = bh % Hc;
  int chunk = blockIdx.y;
  int qblk = blockIdx.x;

  // stage q rows: 32 x 64 halves = 256 x 16B (1 per thread)
  {
    int r = tid >> 3, c = (tid & 7) * 8;
    int srow = (mode == 0) ? (chunk * Wc + qblk * 32 + r)
                           : cls_ids[b * Gc + qblk * 32 + r];
    *(v8h*)&qs[r][c] =
        *(const v8h*)&q[((size_t)b * Sc + srow) * Dc + h * DHc + c];
  }
  if (tid < 32) { mrow[tid] = -1e30f; lrow[tid] = 0.f; }
  __syncthreads();

  v8f o = {};
  int ntiles = (mode == 0) ? 26 : (Sc / 32);
  for (int t = 0; t < ntiles; ++t) {
    // key row index for local key r (or -1 = out of range -> zeros)
    auto key_row = [&](int r) -> int {
      if (mode == 0 && t < 2) return cls_ids[b * Gc + t * 32 + r];
      int pos = (mode == 0) ? (chunk * Wc + (t - 2) * 32 + r - Wc) : (t * 32 + r);
      return (pos >= 0 && pos < Sc) ? pos : -1;
    };
    // stage K tile [key][dh] (b128 loads + stores)
    {
      int r = tid >> 3, c = (tid & 7) * 8;
      int sk = key_row(r);
      v8h kv = {};
      if (sk >= 0)
        kv = *(const v8h*)&k[((size_t)b * Sc + sk) * Dc + h * DHc + c];
      *(v8h*)&ks[r][c] = kv;
    }
    // stage V transposed [dh][key]: key pairs -> packed b32 stores
    {
      int r = (tid >> 4) * 2, c = (tid & 15) * 4;
      int s0 = key_row(r), s1 = key_row(r + 1);
      v4h v0 = {}, v1 = {};
      if (s0 >= 0) v0 = *(const v4h*)&v[((size_t)b * Sc + s0) * Dc + h * DHc + c];
      if (s1 >= 0) v1 = *(const v4h*)&v[((size_t)b * Sc + s1) * Dc + h * DHc + c];
      *(v2h*)&vst[c + 0][r] = (v2h){v0[0], v1[0]};
      *(v2h*)&vst[c + 1][r] = (v2h){v0[1], v1[1]};
      *(v2h*)&vst[c + 2][r] = (v2h){v0[2], v1[2]};
      *(v2h*)&vst[c + 3][r] = (v2h){v0[3], v1[3]};
    }
    if (tid < 32) {
      int ok;
      if (mode == 0 && t < 2) {
        ok = ((t * 32 + tid) < cls_len[b]) ? 1 : 0;
      } else if (mode == 0) {
        int pos = chunk * Wc + (t - 2) * 32 + tid - Wc;
        ok = (pos >= 0 && pos < Sc && am[b * Sc + pos] > 0 &&
              isg[b * Sc + pos] == 0) ? 1 : 0;
      } else {
        int pos = t * 32 + tid;
        ok = (am[b * Sc + pos] > 0) ? 1 : 0;
      }
      colok[tid] = ok;
    }
    __syncthreads();

    // scores S = q @ k^T : 32x32, waves 0..3 each one 16x16 tile (K=64)
    if (wave < 4) {
      int swm = wave >> 1, swn = wave & 1;
      v8f sacc = {};
      v16h a0 = frag_ld(&qs[0][0], 72, swm * 16, 0, lane);
      v16h b0 = frag_ld(&ks[0][0], 72, swn * 16, 0, lane);
      sacc = __builtin_amdgcn_wmma_f32_16x16x32_f16(false, a0, false, b0,
                                                    (short)0, sacc, false, false);
      v16h a1 = frag_ld(&qs[0][0], 72, swm * 16, 32, lane);
      v16h b1 = frag_ld(&ks[0][0], 72, swn * 16, 32, lane);
      sacc = __builtin_amdgcn_wmma_f32_16x16x32_f16(false, a1, false, b1,
                                                    (short)0, sacc, false, false);
      int sub = (lane >> 4) & 1, rl = lane & 15;
#pragma unroll
      for (int j = 0; j < 8; ++j)
        sc[swm * 16 + sub * 8 + j][swn * 16 + rl] = sacc[j];
    }
    __syncthreads();

    // scale + mask
    for (int i = tid; i < 32 * 32; i += 256) {
      int r = i >> 5, c = i & 31;
      float val = sc[r][c] * 0.125f;  // 1/sqrt(64)
      bool ok = colok[c] != 0;
      if (mode == 0 && t >= 2) {
        int rel = ((t - 2) * 32 + c - Wc) - (qblk * 32 + r);
        ok = ok && (rel >= -Wc && rel <= Wc);
      }
      sc[r][c] = ok ? val : NEGF;
    }
    __syncthreads();

    // per-row online softmax update
    if (tid < 32) {
      float mo = mrow[tid], mt = mo;
      for (int c = 0; c < 32; ++c) mt = fmaxf(mt, sc[tid][c]);
      float cf = __expf(mo - mt);
      float ls = 0.f;
      for (int c = 0; c < 32; ++c) {
        float p = __expf(sc[tid][c] - mt);
        pr[tid][c] = (_Float16)p;
        ls += p;
      }
      mrow[tid] = mt;
      lrow[tid] = lrow[tid] * cf + ls;
      cfrow[tid] = cf;
    }
    __syncthreads();

    // rescale O fragments, accumulate P @ V (each wave one 16x16 of 32x64)
    {
      int wm = wave >> 2, wn = wave & 3;
      int sub = (lane >> 4) & 1;
#pragma unroll
      for (int j = 0; j < 8; ++j) o[j] *= cfrow[wm * 16 + sub * 8 + j];
      v16h pa = frag_ld(&pr[0][0], 40, wm * 16, 0, lane);
      v16h vb = frag_ld(&vst[0][0], 40, wn * 16, 0, lane);
      o = __builtin_amdgcn_wmma_f32_16x16x32_f16(false, pa, false, vb,
                                                 (short)0, o, false, false);
    }
    __syncthreads();
  }

  // finalize + write f16 (head-merged [.., D] layout)
  {
    int wm = wave >> 2, wn = wave & 3;
    int sub = (lane >> 4) & 1, rl = lane & 15;
#pragma unroll
    for (int j = 0; j < 8; ++j) {
      int qr = wm * 16 + sub * 8 + j;
      float resv = o[j] / lrow[qr];
      int dc = wn * 16 + rl;
      if (mode == 0) {
        int srow = chunk * Wc + qblk * 32 + qr;
        outp[((size_t)b * Sc + srow) * Dc + h * DHc + dc] = (_Float16)resv;
      } else {
        int g = qblk * 32 + qr;
        outp[((size_t)b * Gc + g) * Dc + h * DHc + dc] = (_Float16)resv;
      }
    }
  }
}

// ---------------- scatter global-query rows into ctx (f16) ----------------
__global__ void scatter_outg_kernel(const _Float16* outg, const int* cls_ids,
                                    const int* cls_len, _Float16* ctx) {
  int bg = blockIdx.x, b = bg / Gc, g = bg % Gc;
  if (g >= cls_len[b]) return;
  int s = cls_ids[bg];
  for (int i = threadIdx.x; i < Dc; i += blockDim.x)
    ctx[((size_t)b * Sc + s) * Dc + i] = outg[(size_t)bg * Dc + i];
}

// ---------------- final gather of valid CLS rows (f32) ----------------
__global__ void gather_cls_kernel(const float* x, const int* cls_ids,
                                  const int* cls_len, float* out) {
  int bg = blockIdx.x, b = bg / Gc, g = bg % Gc;
  int s = cls_ids[bg];
  bool valid = g < cls_len[b];
  for (int i = threadIdx.x; i < Dc; i += blockDim.x)
    out[(size_t)bg * Dc + i] = valid ? x[((size_t)b * Sc + s) * Dc + i] : 0.0f;
}

// ---------------- host orchestration ----------------
extern "C" void kernel_launch(void* const* d_in, const int* in_sizes, int n_in,
                              void* d_out, int out_size, void* d_ws,
                              size_t ws_size, hipStream_t stream) {
  (void)in_sizes; (void)n_in; (void)out_size; (void)ws_size;
  const int*   input_ids  = (const int*)d_in[0];
  const int*   token_type = (const int*)d_in[1];
  const int*   am         = (const int*)d_in[2];
  const int*   cls_ids    = (const int*)d_in[3];
  const int*   cls_len    = (const int*)d_in[4];
  /* d_in[5] = part (unused by reference math) */
  const float* word_emb = (const float*)d_in[6];
  const float* pos_emb  = (const float*)d_in[7];
  const float* type_emb = (const float*)d_in[8];
  const float* eln_s    = (const float*)d_in[9];
  const float* eln_b    = (const float*)d_in[10];
  const float* wq = (const float*)d_in[11];
  const float* bq = (const float*)d_in[12];
  const float* wk = (const float*)d_in[13];
  const float* bk = (const float*)d_in[14];
  const float* wv = (const float*)d_in[15];
  const float* bv = (const float*)d_in[16];
  const float* wo = (const float*)d_in[17];
  const float* bo = (const float*)d_in[18];
  const float* ln1s = (const float*)d_in[19];
  const float* ln1b = (const float*)d_in[20];
  const float* w1 = (const float*)d_in[21];
  const float* b1 = (const float*)d_in[22];
  const float* w2 = (const float*)d_in[23];
  const float* b2 = (const float*)d_in[24];
  const float* ln2s = (const float*)d_in[25];
  const float* ln2b = (const float*)d_in[26];

  const size_t NT = (size_t)Bc * Sc;  // 8192 token rows

  // workspace carve-up (all sub-buffer sizes are multiples of 16 bytes)
  char* p = (char*)d_ws;
  auto alloc_f32 = [&](size_t n) { float* r = (float*)p;    p += n * 4; return r; };
  auto alloc_f16 = [&](size_t n) { _Float16* r = (_Float16*)p; p += n * 2; return r; };
  float*    X    = alloc_f32(NT * Dc);        // f32 residual stream
  float*    Y    = alloc_f32(NT * Dc);        // f32 pre-LN buffer
  _Float16* Xh   = alloc_f16(NT * Dc);        // f16 activations
  _Float16* Qh   = alloc_f16(NT * Dc);
  _Float16* Kh   = alloc_f16(NT * Dc);
  _Float16* Vh   = alloc_f16(NT * Dc);
  _Float16* CTXh = alloc_f16(NT * Dc);
  _Float16* Hbh  = alloc_f16(NT * FFc);
  _Float16* OUTG = alloc_f16((size_t)Bc * Gc * Dc);
  _Float16* WT[Lc][6];                        // per-layer transposed f16 weights
  for (int l = 0; l < Lc; ++l) {
    WT[l][0] = alloc_f16((size_t)Dc * Dc);    // wq^T
    WT[l][1] = alloc_f16((size_t)Dc * Dc);    // wk^T
    WT[l][2] = alloc_f16((size_t)Dc * Dc);    // wv^T
    WT[l][3] = alloc_f16((size_t)Dc * Dc);    // wo^T
    WT[l][4] = alloc_f16((size_t)Dc * FFc);   // w1^T  [FF][D]
    WT[l][5] = alloc_f16((size_t)FFc * Dc);   // w2^T  [D][FF]
  }
  int* ISG = (int*)p;                          // [B*S]

  hipMemsetAsync(ISG, 0, NT * sizeof(int), stream);
  isglob_kernel<<<(Bc * Gc + 255) / 256, 256, 0, stream>>>(cls_ids, cls_len, ISG);
  embed_kernel<<<(unsigned)NT, 256, 0, stream>>>(input_ids, token_type, word_emb,
                                                 pos_emb, type_emb, eln_s, eln_b,
                                                 X, Xh);
  // weight convert+transpose (once per launch)
  for (int l = 0; l < Lc; ++l) {
    const size_t dd = (size_t)Dc * Dc, df = (size_t)Dc * FFc;
    unsigned gdd = (unsigned)((dd + 255) / 256), gdf = (unsigned)((df + 255) / 256);
    convert_transpose_kernel<<<gdd, 256, 0, stream>>>(wq + l * dd, WT[l][0], Dc, Dc);
    convert_transpose_kernel<<<gdd, 256, 0, stream>>>(wk + l * dd, WT[l][1], Dc, Dc);
    convert_transpose_kernel<<<gdd, 256, 0, stream>>>(wv + l * dd, WT[l][2], Dc, Dc);
    convert_transpose_kernel<<<gdd, 256, 0, stream>>>(wo + l * dd, WT[l][3], Dc, Dc);
    convert_transpose_kernel<<<gdf, 256, 0, stream>>>(w1 + l * df, WT[l][4], Dc, FFc);
    convert_transpose_kernel<<<gdf, 256, 0, stream>>>(w2 + l * df, WT[l][5], FFc, Dc);
  }

  dim3 gD(Dc / 128, (unsigned)(NT / 64));
  dim3 gF(FFc / 128, (unsigned)(NT / 64));
  for (int l = 0; l < Lc; ++l) {
    const float* bql = bq + (size_t)l * Dc;
    const float* bkl = bk + (size_t)l * Dc;
    const float* bvl = bv + (size_t)l * Dc;
    const float* bol = bo + (size_t)l * Dc;
    const float* b1l = b1 + (size_t)l * FFc;
    const float* b2l = b2 + (size_t)l * Dc;
    const float* l1s = ln1s + (size_t)l * Dc;
    const float* l1b = ln1b + (size_t)l * Dc;
    const float* l2s = ln2s + (size_t)l * Dc;
    const float* l2b = ln2b + (size_t)l * Dc;

    // QKV projections (f16 out)
    wmma_gemm_kernel<<<gD, 256, 0, stream>>>(Xh, WT[l][0], bql, nullptr, nullptr, Qh, (int)NT, Dc, Dc, 0);
    wmma_gemm_kernel<<<gD, 256, 0, stream>>>(Xh, WT[l][1], bkl, nullptr, nullptr, Kh, (int)NT, Dc, Dc, 0);
    wmma_gemm_kernel<<<gD, 256, 0, stream>>>(Xh, WT[l][2], bvl, nullptr, nullptr, Vh, (int)NT, Dc, Dc, 0);

    // band + global-key attention (one softmax over [sg, band]) -> CTXh
    flash_attn_kernel<<<dim3(Wc / 32, Sc / Wc, Bc * Hc), 256, 0, stream>>>(
        Qh, Kh, Vh, am, ISG, cls_ids, cls_len, CTXh, 0);
    // global-query attention over all S -> OUTG
    flash_attn_kernel<<<dim3(Gc / 32, 1, Bc * Hc), 256, 0, stream>>>(
        Qh, Kh, Vh, am, ISG, cls_ids, cls_len, OUTG, 1);
    scatter_outg_kernel<<<Bc * Gc, 256, 0, stream>>>(OUTG, cls_ids, cls_len, CTXh);

    // output projection + residual -> Y, then LN1 -> X/Xh
    wmma_gemm_kernel<<<gD, 256, 0, stream>>>(CTXh, WT[l][3], bol, X, Y, nullptr, (int)NT, Dc, Dc, 0);
    ln_kernel<<<(unsigned)NT, 256, 0, stream>>>(Y, l1s, l1b, X, Xh);

    // FFN: gelu(X @ w1 + b1) -> Hbh (f16); Hbh @ w2 + b2 + X -> Y; LN2 -> X/Xh
    wmma_gemm_kernel<<<gF, 256, 0, stream>>>(Xh, WT[l][4], b1l, nullptr, nullptr, Hbh, (int)NT, FFc, Dc, 1);
    wmma_gemm_kernel<<<gD, 256, 0, stream>>>(Hbh, WT[l][5], b2l, X, Y, nullptr, (int)NT, Dc, FFc, 0);
    ln_kernel<<<(unsigned)NT, 256, 0, stream>>>(Y, l2s, l2b, X, Xh);
  }

  gather_cls_kernel<<<Bc * Gc, 256, 0, stream>>>(X, cls_ids, cls_len, (float*)d_out);
}
